// CrossAttention_88837103550791
// MI455X (gfx1250) — matrix operand order repaired and checked by reading, not compiled, not run on hardware
//
#include <hip/hip_runtime.h>
#include <hip/hip_bf16.h>

// ---------------------------------------------------------------------------
// CrossAttention for MI455X (gfx1250): f16 WMMA everywhere, f32 accumulate.
// B=8, C=INNER=384, N=1024, HEADS=6, DIM_HEAD=64.
// ---------------------------------------------------------------------------

typedef __attribute__((ext_vector_type(16))) _Float16 v16h;
typedef __attribute__((ext_vector_type(8)))  _Float16 v8h;
typedef __attribute__((ext_vector_type(8)))  float    v8f;

#define NB   8
#define NC   384
#define NN   1024
#define NH   6
#define DH   64
#define WSZ  (NC * NC)       // 147456 weights per matrix

#define WMMA16(A, B, C) \
  __builtin_amdgcn_wmma_f32_16x16x32_f16(false, (A), false, (B), (short)0, (C), false, false)

// ---- fragment loaders (layouts per CDNA5 ISA 7.12.2, wave32) ---------------
// A 16x32 f16: lane = row (lane&15); K elements: e<8 -> k0+e, e>=8 -> k0+8+e,
//              with k0 = (lane>>4)*8.  Two 16B loads.
__device__ __forceinline__ v16h load_frag_a(const _Float16* __restrict__ base, int ld) {
  const int lane = threadIdx.x & 31;
  const _Float16* p = base + (lane & 15) * ld + ((lane >> 4) * 8);
  v8h lo = *(const v8h*)(p);
  v8h hi = *(const v8h*)(p + 16);
  v16h f;
#pragma unroll
  for (int i = 0; i < 8; ++i) { f[i] = lo[i]; f[i + 8] = hi[i]; }
  return f;
}

// B 32x16 f16: lane = column (lane&15); K elements: e<8 -> k0+e, e>=8 -> k0+8+e,
//              with k0 = (lane>>4)*8 (same as A).
__device__ __forceinline__ v16h load_frag_b(const _Float16* __restrict__ base, int ld) {
  const int lane = threadIdx.x & 31;
  const _Float16* p = base + (lane & 15) * ld + ((lane >> 4) * 8);
  v8h lo = *(const v8h*)(p);
  v8h hi = *(const v8h*)(p + 16);
  v16h f;
#pragma unroll
  for (int i = 0; i < 8; ++i) { f[i] = lo[i]; f[i + 8] = hi[i]; }
  return f;
}

// 16-lane (intra-half) reductions: a C/D-tile row lives on 16 lanes of a half.
__device__ __forceinline__ float half16_max(float v) {
  v = fmaxf(v, __shfl_xor(v, 1, 32));
  v = fmaxf(v, __shfl_xor(v, 2, 32));
  v = fmaxf(v, __shfl_xor(v, 4, 32));
  v = fmaxf(v, __shfl_xor(v, 8, 32));
  return v;
}
__device__ __forceinline__ float half16_sum(float v) {
  v += __shfl_xor(v, 1, 32);
  v += __shfl_xor(v, 2, 32);
  v += __shfl_xor(v, 4, 32);
  v += __shfl_xor(v, 8, 32);
  return v;
}

// ---------------------------------------------------------------------------
// Kernel W: convert weights f32 -> f16, zero the attn_map output region.
// ---------------------------------------------------------------------------
__global__ void ca_wconv(const float* __restrict__ wq, const float* __restrict__ wk,
                         const float* __restrict__ wv, const float* __restrict__ wo,
                         _Float16* __restrict__ w16, float* __restrict__ attnmap) {
  const int i = blockIdx.x * blockDim.x + threadIdx.x;
  if (i < WSZ) {
    w16[i]           = (_Float16)wq[i];
    w16[WSZ + i]     = (_Float16)wk[i];
    w16[2 * WSZ + i] = (_Float16)wv[i];
    w16[3 * WSZ + i] = (_Float16)wo[i];
  }
  if (i < NB * NN) attnmap[i] = 0.0f;
}

// ---------------------------------------------------------------------------
// Kernel A: QKV projections.  grid = (B, 16 n-blocks of 64, {q,k,v}).
// Y[o,n] = sum_c W[o,c] X[c,n].  Input block staged transposed in LDS (f16).
// Each wave owns an o-tile: 12 A-fragments preloaded once (24 clustered
// b128 loads), reused across 4 n-tiles of LDS B-fragments.
// q/k stored token-major [b][h][n][64]; v stored d-major [b][o][n].
// ---------------------------------------------------------------------------
__global__ __launch_bounds__(256) void ca_proj_qkv(
    const float* __restrict__ query, const float* __restrict__ context,
    const _Float16* __restrict__ w16,
    _Float16* __restrict__ q16, _Float16* __restrict__ k16, _Float16* __restrict__ v16) {
  const int b = blockIdx.x, nb = blockIdx.y, m = blockIdx.z;
  const int n0 = nb * 64;
  __shared__ __align__(16) _Float16 Xt[64][392];   // [token][channel], padded stride

  const float* __restrict__ X = (m == 0 ? query : context) + (size_t)b * NC * NN;
  for (int idx = threadIdx.x; idx < NC * 64; idx += 256) {
    const int c = idx >> 6, nl = idx & 63;
    Xt[nl][c] = (_Float16)X[c * NN + n0 + nl];
  }
  __syncthreads();

  const _Float16* __restrict__ W = w16 + m * WSZ;
  const int wave = threadIdx.x >> 5, lane = threadIdx.x & 31;
  const int half = lane >> 4, cidx = lane & 15;

  for (int ot = wave; ot < 24; ot += 8) {         // 3 o-tiles per wave
    v16h af[12];
#pragma unroll
    for (int cc = 0; cc < 12; ++cc)               // bulk-issued weight loads
      af[cc] = load_frag_a(W + ot * 16 * NC + cc * 32, NC);
#pragma unroll
    for (int nt = 0; nt < 4; ++nt) {
      v8f acc = {};
#pragma unroll
      for (int cc = 0; cc < 12; ++cc) {
        v16h bf = load_frag_b(&Xt[nt * 16][cc * 32], 392);
        acc = WMMA16(af[cc], bf, acc);
      }
      const int n  = n0 + nt * 16 + cidx;
      const int ob = ot * 16 + half * 8;          // first of 8 consecutive o rows
      if (m < 2) {                                // q/k: token-major, one 16B store
        _Float16* dst = (m == 0 ? q16 : k16);
        const int h = ob >> 6, d0 = ob & 63;
        v8h st;
#pragma unroll
        for (int e = 0; e < 8; ++e) st[e] = (_Float16)acc[e];
        *(v8h*)(dst + ((size_t)(b * NH + h) * NN + n) * DH + d0) = st;
      } else {                                    // v: d-major [b][o][n]
#pragma unroll
        for (int e = 0; e < 8; ++e)
          v16[(size_t)(b * NC + ob + e) * NN + n] = (_Float16)acc[e];
      }
    }
  }
}

// ---------------------------------------------------------------------------
// Kernel B: attention.  grid = (48 bh, 8 row-blocks of 128).  8 waves x 16 rows.
// All waves share the same K/V: cooperative LDS staging (K chunk 64x64,
// V chunk 64x32), with global_prefetch of the next chunk.  Two-pass online
// softmax; P transposed via per-wave LDS tile; O = P * V^T.  attn_map column
// sums accumulated with global f32 atomics.
// ---------------------------------------------------------------------------
__global__ __launch_bounds__(256) void ca_attention(
    const _Float16* __restrict__ q16, const _Float16* __restrict__ k16,
    const _Float16* __restrict__ v16, _Float16* __restrict__ o16,
    float* __restrict__ attnmap) {
  const int bh = blockIdx.x;                 // b*6 + h
  const int b = bh / NH, h = bh % NH;
  const int tid = threadIdx.x;
  const int wave = tid >> 5, lane = tid & 31;
  const int half = lane >> 4, cidx = lane & 15;
  const int n0 = blockIdx.y * 128 + wave * 16;
  const float scale = 0.125f;                // 64^-0.5
  const float mapscale = 1.0f / (float)(NH * NN);

  __shared__ __align__(16) _Float16 Kt[64][72];        // K tokens, padded (conflict-free)
  __shared__ __align__(16) _Float16 Vt[64][40];        // V d-major chunk
  __shared__ __align__(16) _Float16 Pbuf[8][16][40];   // per-wave P transpose tile

  const _Float16* __restrict__ qb = q16 + ((size_t)bh * NN + n0) * DH;
  const _Float16* __restrict__ kb = k16 + (size_t)bh * NN * DH;
  const _Float16* __restrict__ vb = v16 + (size_t)(b * NC + h * DH) * NN;

  const v16h qa0 = load_frag_a(qb, DH);        // Q rows, d = 0..31
  const v16h qa1 = load_frag_a(qb + 32, DH);   // Q rows, d = 32..63

  float M[8], L[8];
#pragma unroll
  for (int e = 0; e < 8; ++e) { M[e] = -1e30f; L[e] = 0.0f; }

  // ---- pass 1: per-row running max and exp-sum over all 1024 keys ---------
  for (int kc = 0; kc < 16; ++kc) {            // 16 chunks of 64 keys
#pragma unroll
    for (int r = 0; r < 2; ++r) {              // stage 64 tok x 64 d (8 KB)
      const int flat = (tid + r * 256) * 8;
      const int tok = flat >> 6, d0 = flat & 63;
      *(v8h*)(&Kt[tok][d0]) = *(const v8h*)(kb + (size_t)(kc * 64 + tok) * DH + d0);
    }
    if (kc < 15)                                // prefetch next K chunk
      __builtin_prefetch(kb + (size_t)(kc + 1) * 64 * DH + tid * 16, 0, 3);
    __syncthreads();
#pragma unroll
    for (int sub = 0; sub < 4; ++sub) {
      v16h b0 = load_frag_b(&Kt[sub * 16][0], 72);
      v16h b1 = load_frag_b(&Kt[sub * 16][32], 72);
      v8f s = {};
      s = WMMA16(qa0, b0, s);
      s = WMMA16(qa1, b1, s);
#pragma unroll
      for (int e = 0; e < 8; ++e) {
        const float sv = s[e] * scale;
        const float tmax = half16_max(sv);
        const float nm = fmaxf(M[e], tmax);
        const float ssum = half16_sum(__expf(sv - nm));
        L[e] = L[e] * __expf(M[e] - nm) + ssum;
        M[e] = nm;
      }
    }
    __syncthreads();
  }
  float invL[8];
#pragma unroll
  for (int e = 0; e < 8; ++e) invL[e] = 1.0f / L[e];

  // ---- pass 2: exact P, column sums (attn_map), O = P * V^T ---------------
  v8f o[4] = {{}, {}, {}, {}};
  for (int mc = 0; mc < 32; ++mc) {            // 32-wide m chunks
    {                                          // stage K 32x64 and V 64x32
      const int flat = tid * 8;
      const int tok = flat >> 6, d0 = flat & 63;
      *(v8h*)(&Kt[tok][d0]) = *(const v8h*)(kb + (size_t)(mc * 32 + tok) * DH + d0);
      const int dv = flat >> 5, ml = flat & 31;
      *(v8h*)(&Vt[dv][ml]) = *(const v8h*)(vb + (size_t)dv * NN + mc * 32 + ml);
    }
    if (mc < 31) {
      __builtin_prefetch(kb + (size_t)(mc + 1) * 32 * DH + tid * 16, 0, 3);
      __builtin_prefetch(vb + (size_t)(tid >> 2) * NN + (mc + 1) * 32 + (tid & 3) * 16, 0, 3);
    }
    __syncthreads();
#pragma unroll
    for (int sub = 0; sub < 2; ++sub) {
      v16h b0 = load_frag_b(&Kt[sub * 16][0], 72);
      v16h b1 = load_frag_b(&Kt[sub * 16][32], 72);
      v8f s = {};
      s = WMMA16(qa0, b0, s);
      s = WMMA16(qa1, b1, s);
      float csum = 0.0f;
#pragma unroll
      for (int e = 0; e < 8; ++e) {
        const float p = __expf(s[e] * scale - M[e]) * invL[e];
        csum += p;
        Pbuf[wave][e + half * 8][sub * 16 + cidx] = (_Float16)p;
      }
      csum += __shfl_xor(csum, 16, 32);         // combine the two row halves
      if (lane < 16)
        atomicAdd(&attnmap[b * NN + mc * 32 + sub * 16 + lane], csum * mapscale);
    }
    const v16h pa = load_frag_a(&Pbuf[wave][0][0], 40);
#pragma unroll
    for (int dt = 0; dt < 4; ++dt) {
      v16h vf = load_frag_b(&Vt[dt * 16][0], 40);
      o[dt] = WMMA16(pa, vf, o[dt]);
    }
    __syncthreads();
  }

  // ---- store O token-major [b][n][inner] (f16) ----------------------------
  _Float16* __restrict__ od = o16 + ((size_t)b * NN + n0) * NC + h * DH;
#pragma unroll
  for (int dt = 0; dt < 4; ++dt)
#pragma unroll
    for (int e = 0; e < 8; ++e)
      od[(e + half * 8) * NC + dt * 16 + cidx] = (_Float16)o[dt][e];
}

// ---------------------------------------------------------------------------
// Kernel C: output projection out = Wo * O + bo.  grid = (B, 16 n-blocks).
// O block staged in LDS once per workgroup; Wo A-fragments preloaded per wave.
// ---------------------------------------------------------------------------
__global__ __launch_bounds__(256) void ca_proj_out(
    const _Float16* __restrict__ wo16, const _Float16* __restrict__ o16,
    const float* __restrict__ bo, float* __restrict__ out) {
  const int b = blockIdx.x, nb = blockIdx.y;
  const int n0 = nb * 64;
  const int wave = threadIdx.x >> 5, lane = threadIdx.x & 31;
  const int half = lane >> 4, cidx = lane & 15;

  __shared__ __align__(16) _Float16 Ot[64][392];   // [token][inner], padded stride
  const _Float16* __restrict__ Osrc = o16 + ((size_t)b * NN + n0) * NC;
  for (int idx = threadIdx.x * 8; idx < 64 * NC; idx += 256 * 8) {
    const int tok = idx / NC, c = idx % NC;
    *(v8h*)(&Ot[tok][c]) = *(const v8h*)(Osrc + tok * NC + c);
  }
  __syncthreads();

  for (int ot = wave; ot < 24; ot += 8) {
    v16h af[12];
#pragma unroll
    for (int cc = 0; cc < 12; ++cc)
      af[cc] = load_frag_a(wo16 + ot * 16 * NC + cc * 32, NC);
#pragma unroll
    for (int nt = 0; nt < 4; ++nt) {
      v8f acc = {};
#pragma unroll
      for (int cc = 0; cc < 12; ++cc) {
        v16h bf = load_frag_b(&Ot[nt * 16][cc * 32], 392);
        acc = WMMA16(af[cc], bf, acc);
      }
      const int n  = n0 + nt * 16 + cidx;
      const int ob = ot * 16 + half * 8;
#pragma unroll
      for (int e = 0; e < 8; ++e)
        out[(size_t)(b * NC + ob + e) * NN + n] = acc[e] + bo[ob + e];
    }
  }
}

// ---------------------------------------------------------------------------
// Launch
// ---------------------------------------------------------------------------
extern "C" void kernel_launch(void* const* d_in, const int* in_sizes, int n_in,
                              void* d_out, int out_size, void* d_ws, size_t ws_size,
                              hipStream_t stream) {
  const float* query   = (const float*)d_in[0];
  const float* context = (const float*)d_in[1];
  const float* Wq = (const float*)d_in[2];
  const float* Wk = (const float*)d_in[3];
  const float* Wv = (const float*)d_in[4];
  const float* Wo = (const float*)d_in[5];
  const float* bo = (const float*)d_in[6];

  float* out     = (float*)d_out;                   // [8][384][1024]
  float* attnmap = out + (size_t)NB * NC * NN;      // [8][1024]

  // workspace layout (all f16, 16B-aligned), total ~25.2 MB
  char* ws = (char*)d_ws;
  _Float16* w16  = (_Float16*)(ws);                 // Wq,Wk,Wv,Wo f16 (4*294912 B)
  _Float16* wo16 = (_Float16*)(ws + 884736);        // alias of 4th weight slot
  _Float16* q16  = (_Float16*)(ws + 1179648);       // [8][6][1024][64]
  _Float16* k16  = (_Float16*)(ws + 7471104);       // [8][6][1024][64]
  _Float16* v16  = (_Float16*)(ws + 13762560);      // [8][384][1024]
  _Float16* o16  = (_Float16*)(ws + 20054016);      // [8][1024][384]

  ca_wconv<<<(WSZ + 255) / 256, 256, 0, stream>>>(Wq, Wk, Wv, Wo, w16, attnmap);
  ca_proj_qkv<<<dim3(NB, 16, 3), 256, 0, stream>>>(query, context, w16, q16, k16, v16);
  ca_attention<<<dim3(NB * NH, 8), 256, 0, stream>>>(q16, k16, v16, o16, attnmap);
  ca_proj_out<<<dim3(NB, 16), 256, 0, stream>>>(wo16, o16, bo, out);
}
